// LearningModel_43190191128869
// MI455X (gfx1250) — matrix operand order repaired and checked by reading, not compiled, not run on hardware
//
#include <hip/hip_runtime.h>
#include <hip/hip_bf16.h>
#include <math.h>

// ---------------------------------------------------------------------------
// Problem constants (match reference)
// ---------------------------------------------------------------------------
#define DIM     256
#define HID     256
#define NRULE   8
#define NINIT   8192
#define LAYERS  16
#define MM      4096
#define NTOTAL  (NINIT + LAYERS * MM)
// worst-case 16-aligned rule buckets: 4096/16 + 8 pad tiles
#define MAXT    264
#define EVAL_BLOCKS (NTOTAL / 256)   // 288

typedef _Float16 v16h __attribute__((ext_vector_type(16)));
typedef _Float16 v8h  __attribute__((ext_vector_type(8)));
typedef float    v8f  __attribute__((ext_vector_type(8)));

// ---------------------------------------------------------------------------
// Kernel: initial embeddings  store[i] = init_table[thax[i]] + sine_table[sine[i]]
// stored as f16 (WMMA A-operand dtype)
// ---------------------------------------------------------------------------
__global__ void k_init(const float* __restrict__ init_tab,
                       const float* __restrict__ sine_tab,
                       const int* __restrict__ thax,
                       const int* __restrict__ sine,
                       _Float16* __restrict__ store)
{
    int i = blockIdx.x * blockDim.x + threadIdx.x;   // over NINIT*DIM
    int node = i >> 8;
    int d    = i & 255;
    float v = init_tab[(size_t)thax[node] * DIM + d] + sine_tab[(size_t)sine[node] * DIM + d];
    store[i] = (_Float16)v;
}

// ---------------------------------------------------------------------------
// Weight pre-swizzle into per-lane-contiguous WMMA B fragments (f16).
// B (KxN=32x16) layout: lanes 0-15 hold K {0-7,16-23}, lanes 16-31 hold
// K {8-15,24-31}, column N = lane%16.  One lane's fragment = 16 halves = 32 B.
// Fragment index t = ((r*NT + nt)*NKB + kb)*32 + lane, stored at t*16 halves.
// ---------------------------------------------------------------------------
__global__ void k_swz1(const float* __restrict__ W1, _Float16* __restrict__ W1s)
{
    int t = blockIdx.x * blockDim.x + threadIdx.x;   // NRULE*16*16*32 = 65536
    int lane = t & 31;
    int kb   = (t >> 5) & 15;
    int nt   = (t >> 9) & 15;
    int r    = t >> 13;
    int n      = nt * 16 + (lane & 15);
    int kstart = kb * 32 + ((lane < 16) ? 0 : 8);
    _Float16* dst = W1s + (size_t)t * 16;
    const float* src = W1 + ((size_t)r * 512 + kstart) * DIM + n;
#pragma unroll
    for (int h = 0; h < 16; ++h) dst[h] = (_Float16)src[(size_t)(h < 8 ? h : h + 8) * DIM];
}

__global__ void k_swz2(const float* __restrict__ W2, _Float16* __restrict__ W2s)
{
    int t = blockIdx.x * blockDim.x + threadIdx.x;   // NRULE*16*8*32 = 32768
    int lane = t & 31;
    int kb   = (t >> 5) & 7;
    int nt   = (t >> 8) & 15;
    int r    = t >> 12;
    int n      = nt * 16 + (lane & 15);
    int kstart = kb * 32 + ((lane < 16) ? 0 : 8);
    _Float16* dst = W2s + (size_t)t * 16;
    const float* src = W2 + ((size_t)r * HID + kstart) * DIM + n;
#pragma unroll
    for (int h = 0; h < 16; ++h) dst[h] = (_Float16)src[(size_t)(h < 8 ? h : h + 8) * DIM];
}

// ---------------------------------------------------------------------------
// Per-layer rule bucketing (single block).  Buckets are 16-aligned; padding
// slots hold -1.  tileRule[t] = rule of tile t, or -1 for unused tiles.
// ---------------------------------------------------------------------------
__global__ void k_sort(const int* __restrict__ rule_row,
                       int* __restrict__ sortedIdx,
                       int* __restrict__ tileRule)
{
    __shared__ int cnt[NRULE];
    __shared__ int cursor[NRULE];
    int tid = threadIdx.x;                      // 256 threads
    if (tid < NRULE) cnt[tid] = 0;
    __syncthreads();
    for (int m = tid; m < MM; m += 256) atomicAdd(&cnt[rule_row[m]], 1);
    __syncthreads();
    if (tid == 0) {
        int cum = 0, t = 0;
        for (int r = 0; r < NRULE; ++r) {
            cursor[r] = cum;
            int ntl = (cnt[r] + 15) >> 4;
            for (int i = 0; i < ntl; ++i) tileRule[t++] = r;
            cum += ntl * 16;
        }
        while (t < MAXT) tileRule[t++] = -1;
    }
    __syncthreads();
    for (int i = tid; i < MAXT * 16; i += 256) sortedIdx[i] = -1;
    __syncthreads();
    for (int m = tid; m < MM; m += 256) {
        int r = rule_row[m];
        int p = atomicAdd(&cursor[r], 1);
        sortedIdx[p] = m;    // order within bucket irrelevant: each node owns its row
    }
}

// ---------------------------------------------------------------------------
// Per-layer GEMM: one workgroup (4 waves) per 16-node tile, single rule/tile.
// N-dimension split across the 4 waves (4 N-tiles each) -> 1056 concurrent
// waves per layer instead of 264, same B-fragment L2 traffic per block.
// Layer1: X[16x512](f16) @ W1[r][512x256] -> ReLU(+b1+tw*t1) -> hbuf (LDS f16)
// Layer2: h[16x256] @ W2[r][256x256] (+b2) -> store rows (f16)
// f16 A-operand layout: row = lane%16; lanes<16 hold K {0-7,16-23} of each
// 32-wide K block, lanes>=16 hold K {8-15,24-31}.
// ---------------------------------------------------------------------------
__global__ __launch_bounds__(128) void k_layer(
    const _Float16* store, _Float16* storeOut,
    const _Float16* __restrict__ W1s, const _Float16* __restrict__ W2s,
    const float* __restrict__ b1, const float* __restrict__ t1,
    const float* __restrict__ b2, const float* __restrict__ tweaks,
    const int* __restrict__ parents, const int* __restrict__ sortedIdx,
    const int* __restrict__ tileRule, int baseNode)
{
    int tile = blockIdx.x;
    int rule = tileRule[tile];
    if (rule < 0) return;                 // uniform across block -> EXEC stays full

    int wave = threadIdx.x >> 5;          // 0..3, owns N-tiles [wave*4, wave*4+4)
    int lane = threadIdx.x & 31;
    int hiH  = (lane >= 16) ? 1 : 0;
    int ln   = lane & 15;

    int m  = sortedIdx[tile * 16 + ln];
    int mv = (m < 0) ? 0 : m;             // padding lanes compute garbage, never stored
    int p0 = parents[2 * mv + 0];
    int p1 = parents[2 * mv + 1];
    const _Float16* rowP0 = store + (size_t)p0 * DIM;
    const _Float16* rowP1 = store + (size_t)p1 * DIM;

    float tw = tweaks[1];                 // prob_tweaks[0,1] -> deriv tweak

    __shared__ __align__(16) _Float16 hbuf[16 * HID];

    v8f acc[4];
    v8f zero = {};
#pragma unroll
    for (int q = 0; q < 4; ++q) acc[q] = zero;

    // ---- layer 1: K = 512 (2 parents x 256), 16 K-blocks of 32 ----
    for (int kb = 0; kb < 16; ++kb) {
        int koff = kb * 32 + (hiH ? 8 : 0);
        const _Float16* src = (kb < 8) ? (rowP0 + koff) : (rowP1 + (koff - DIM));
        v8h a0 = *(const v8h*)(src);
        v8h a1 = *(const v8h*)(src + 16);
        v16h A;
#pragma unroll
        for (int i = 0; i < 8; ++i) { A[i] = a0[i]; A[i + 8] = a1[i]; }
        const _Float16* bbase = W1s + (size_t)rule * (16 * 16 * 512)
                                    + (size_t)(wave * 4) * (16 * 512)
                                    + (size_t)kb * 512 + (size_t)lane * 16;
#pragma unroll
        for (int q = 0; q < 4; ++q) {
            v16h B = *(const v16h*)(bbase + (size_t)q * (16 * 512));
            acc[q] = __builtin_amdgcn_wmma_f32_16x16x32_f16(
                false, A, false, B, (short)0, acc[q], false, false);
        }
    }

    // ---- bias + tweak + ReLU, stage hidden through LDS ----
#pragma unroll
    for (int q = 0; q < 4; ++q) {
        int col  = (wave * 4 + q) * 16 + ln;
        float bb = b1[(size_t)rule * HID + col] + tw * t1[(size_t)rule * HID + col];
#pragma unroll
        for (int j = 0; j < 8; ++j) {
            float v = acc[q][j] + bb;     // C layout: row = j + hiH*8, col
            v = v > 0.0f ? v : 0.0f;
            hbuf[(j + hiH * 8) * HID + col] = (_Float16)v;
        }
    }
    __syncthreads();                      // all 256 hidden cols visible to all waves

    // ---- layer 2: K = 256, 8 K-blocks of 32 ----
#pragma unroll
    for (int q = 0; q < 4; ++q) acc[q] = zero;
    for (int kb = 0; kb < 8; ++kb) {
        int koff = kb * 32 + (hiH ? 8 : 0);
        const _Float16* src = hbuf + ln * HID + koff;
        v8h a0 = *(const v8h*)(src);
        v8h a1 = *(const v8h*)(src + 16);
        v16h A;
#pragma unroll
        for (int i = 0; i < 8; ++i) { A[i] = a0[i]; A[i + 8] = a1[i]; }
        const _Float16* bbase = W2s + (size_t)rule * (16 * 8 * 512)
                                    + (size_t)(wave * 4) * (8 * 512)
                                    + (size_t)kb * 512 + (size_t)lane * 16;
#pragma unroll
        for (int q = 0; q < 4; ++q) {
            v16h B = *(const v16h*)(bbase + (size_t)q * (8 * 512));
            acc[q] = __builtin_amdgcn_wmma_f32_16x16x32_f16(
                false, A, false, B, (short)0, acc[q], false, false);
        }
    }

    // ---- +b2, scatter rows back to store (each wave owns its 64 columns) ----
#pragma unroll
    for (int q = 0; q < 4; ++q) {
        int col  = (wave * 4 + q) * 16 + ln;
        float bb = b2[(size_t)rule * DIM + col];
#pragma unroll
        for (int j = 0; j < 8; ++j) {
            int row = j + hiH * 8;
            int mm  = sortedIdx[tile * 16 + row];
            if (mm >= 0)
                storeOut[(size_t)(baseNode + mm) * DIM + col] = (_Float16)(acc[q][j] + bb);
        }
    }
}

// ---------------------------------------------------------------------------
// Eval + per-block partial reduction (deterministic tree reduce, no FP atomics).
// Per node: v = <emb, w_eval> + b_eval + tw*t_eval, then six weighted sums.
// loss = pw*SA + SB with SA = sum lab*tot*target*(-ls), SB = sum lab*tot*(1-target)*(-lns)
// ---------------------------------------------------------------------------
__global__ void k_eval(const _Float16* __restrict__ store,
                       const float* __restrict__ w_eval,
                       const float* __restrict__ b_eval,
                       const float* __restrict__ t_eval,
                       const float* __restrict__ tweaks,
                       const float* __restrict__ pos,
                       const float* __restrict__ neg,
                       const int* __restrict__ labeled,
                       float* __restrict__ partials)
{
    int i = blockIdx.x * 256 + threadIdx.x;
    const _Float16* row = store + (size_t)i * DIM;
    float s = 0.0f;
    for (int d = 0; d < DIM; d += 8) {
        v8h x = *(const v8h*)(row + d);
#pragma unroll
        for (int j = 0; j < 8; ++j) s += (float)x[j] * w_eval[d + j];
    }
    float v = s + b_eval[0] + tweaks[0] * t_eval[0];

    float lab = (float)labeled[i];
    float p   = pos[i];
    float ng  = neg[i];
    float tot = p + ng;
    float target = p / fmaxf(tot, 1e-8f);
    float l1p = log1pf(expf(-fabsf(v)));
    float ls  = fminf(v, 0.0f) - l1p;     // log_sigmoid(v)
    float lns = fminf(-v, 0.0f) - l1p;    // log_sigmoid(-v)

    float vals6[6];
    vals6[0] = lab * tot * target * (-ls);          // SA (pw-weighted part)
    vals6[1] = lab * tot * (1.0f - target) * (-lns);// SB
    vals6[2] = lab * p  * ((v >= 0.0f) ? 1.0f : 0.0f);  // posOK
    vals6[3] = lab * ng * ((v <  0.0f) ? 1.0f : 0.0f);  // negOK
    vals6[4] = p  * lab;                             // tot_pos
    vals6[5] = ng * lab;                             // tot_neg

    __shared__ float red[256];
    for (int k = 0; k < 6; ++k) {
        red[threadIdx.x] = vals6[k];
        __syncthreads();
        for (int s2 = 128; s2 > 0; s2 >>= 1) {
            if (threadIdx.x < s2) red[threadIdx.x] += red[threadIdx.x + s2];
            __syncthreads();
        }
        if (threadIdx.x == 0) partials[(size_t)blockIdx.x * 6 + k] = red[0];
        __syncthreads();
    }
}

__global__ void k_final(const float* __restrict__ partials, int nb,
                        float* __restrict__ out)
{
    if (blockIdx.x == 0 && threadIdx.x == 0) {
        float SA = 0, SB = 0, POK = 0, NOK = 0, TP = 0, TN = 0;
        for (int b = 0; b < nb; ++b) {
            SA  += partials[b * 6 + 0];
            SB  += partials[b * 6 + 1];
            POK += partials[b * 6 + 2];
            NOK += partials[b * 6 + 3];
            TP  += partials[b * 6 + 4];
            TN  += partials[b * 6 + 5];
        }
        float pw = TN / fmaxf(TP, 1e-8f);   // POS_WEIGHT_EXTRA = 1.0
        out[0] = pw * SA + SB;
        out[1] = POK;
        out[2] = NOK;
        out[3] = TP;
        out[4] = TN;
    }
}

// ---------------------------------------------------------------------------
// Host launcher
// ---------------------------------------------------------------------------
extern "C" void kernel_launch(void* const* d_in, const int* in_sizes, int n_in,
                              void* d_out, int out_size, void* d_ws, size_t ws_size,
                              hipStream_t stream)
{
    const float* tweaks     = (const float*)d_in[0];
    const float* init_tab   = (const float*)d_in[1];
    const float* sine_tab   = (const float*)d_in[2];
    const float* W1         = (const float*)d_in[3];
    const float* b1         = (const float*)d_in[4];
    const float* t1         = (const float*)d_in[5];
    const float* W2         = (const float*)d_in[6];
    const float* b2         = (const float*)d_in[7];
    const float* w_eval     = (const float*)d_in[8];
    const float* b_eval     = (const float*)d_in[9];
    const float* t_eval     = (const float*)d_in[10];
    const float* pos        = (const float*)d_in[11];
    const float* neg        = (const float*)d_in[12];
    const int*   init_thax  = (const int*)d_in[13];
    const int*   init_sine  = (const int*)d_in[14];
    const int*   deriv_rule = (const int*)d_in[15];
    const int*   deriv_par  = (const int*)d_in[16];
    const int*   labeled    = (const int*)d_in[17];
    float* out = (float*)d_out;
    (void)in_sizes; (void)n_in; (void)out_size; (void)ws_size;

    char* ws = (char*)d_ws;
    size_t off = 0;
    auto carve = [&](size_t bytes) -> void* {
        void* p = ws + off;
        off = (off + bytes + 255) & ~(size_t)255;
        return p;
    };
    _Float16* storeH   = (_Float16*)carve((size_t)NTOTAL * DIM * sizeof(_Float16)); // ~38 MB
    _Float16* W1s      = (_Float16*)carve((size_t)NRULE * 16 * 16 * 512 * sizeof(_Float16)); // 2 MB
    _Float16* W2s      = (_Float16*)carve((size_t)NRULE * 16 * 8 * 512 * sizeof(_Float16));  // 1 MB
    int*      sortedIdx= (int*)carve((size_t)MAXT * 16 * sizeof(int));
    int*      tileRule = (int*)carve((size_t)MAXT * sizeof(int));
    float*    partials = (float*)carve((size_t)EVAL_BLOCKS * 6 * sizeof(float));

    // initial embeddings (f16 store)
    k_init<<<(NINIT * DIM) / 256, 256, 0, stream>>>(init_tab, sine_tab, init_thax,
                                                    init_sine, storeH);
    // one-time weight conversion + fragment swizzle
    k_swz1<<<(NRULE * 16 * 16 * 32) / 256, 256, 0, stream>>>(W1, W1s);
    k_swz2<<<(NRULE * 16 * 8 * 32) / 256, 256, 0, stream>>>(W2, W2s);

    // layered DAG walk
    for (int l = 0; l < LAYERS; ++l) {
        k_sort<<<1, 256, 0, stream>>>(deriv_rule + (size_t)l * MM, sortedIdx, tileRule);
        k_layer<<<MAXT, 128, 0, stream>>>(storeH, storeH, W1s, W2s, b1, t1, b2, tweaks,
                                          deriv_par + (size_t)l * MM * 2,
                                          sortedIdx, tileRule, NINIT + l * MM);
    }

    // eval net + loss reductions
    k_eval<<<EVAL_BLOCKS, 256, 0, stream>>>(storeH, w_eval, b_eval, t_eval, tweaks,
                                            pos, neg, labeled, partials);
    k_final<<<1, 32, 0, stream>>>(partials, EVAL_BLOCKS, out);
}
